// SwinTransformerBlock_3100966388154
// MI455X (gfx1250) — matrix operand order repaired and checked by reading, not compiled, not run on hardware
//
#include <hip/hip_runtime.h>
#include <hip/hip_bf16.h>
#include <math.h>

typedef __attribute__((ext_vector_type(16))) _Float16 v16h;
typedef __attribute__((ext_vector_type(8)))  _Float16 v8h;
typedef __attribute__((ext_vector_type(4)))  _Float16 v4h;
typedef __attribute__((ext_vector_type(8)))  float    v8f;

#define C_INd    256
#define C_INNERd 128
#define N_WIN    4096

// LDS row strides (halfs); all *2 bytes are multiples of 16 for b128 alignment.
#define LDX  264   // X16T [64][264]  (s, c)   K=256
#define LDI  136   // THT/PHT/Yn [64][136] (.,i) K=128
#define LDT  72    // GG [128][72], ATn [64][72] (., t) K=64
#define LDF  65    // F [64][65] f32

static constexpr size_t SMEM_BYTES =
    (size_t)(64 * LDX + 2 * 64 * LDI + 128 * LDT + 64 * LDT + 64 * LDI) * 2 +
    (size_t)(64 * LDF) * 4;   // 130304 bytes

// f16 weight scratch layout in d_ws (halfs):
#define WOFF_TH 0
#define WOFF_PH 32768
#define WOFF_G  65536
#define WOFF_W  98304
#define W_TOTAL 131072   // halfs (262144 bytes)

// Window-flat element j of window n -> flat index into x (B,C=256,T=16,H=128,W=128).
// flat bit layout: [2:0]=wi [5:3]=hi [8:6]=ti [12:9]=w16 [16:13]=h16 [17]=t2 [25:18]=c
__device__ __forceinline__ int win_to_x(int n, int j) {
  int flat = (n << 14) + j;
  int wi  =  flat        & 7;
  int hi  = (flat >> 3)  & 7;
  int ti  = (flat >> 6)  & 7;
  int w16 = (flat >> 9)  & 15;
  int h16 = (flat >> 13) & 15;
  int t2  = (flat >> 17) & 1;
  int c   = (flat >> 18) & 255;
  int t = (t2 << 3) + ti;
  int h = (h16 << 3) + hi;
  int w = (w16 << 3) + wi;
  return ((c * 16 + t) * 128 + h) * 128 + w;
}

__device__ __forceinline__ v8f zero8() {
  v8f z;
#pragma unroll
  for (int i = 0; i < 8; ++i) z[i] = 0.0f;
  return z;
}

__device__ __forceinline__ v16h cat8(v8h a, v8h b) {
  return __builtin_shufflevector(a, b, 0, 1, 2, 3, 4, 5, 6, 7,
                                 8, 9, 10, 11, 12, 13, 14, 15);
}

// A fragment (16x32 f16) from f16 row-major [M x ld] (global or LDS):
// per-lane halves are two contiguous 8-half runs -> 2x b128 loads.
__device__ __forceinline__ v16h load_a16(const _Float16* __restrict__ W, int ld,
                                         int m0, int k0, int lane) {
  const _Float16* row = W + (m0 + (lane & 15)) * ld + k0 + ((lane >> 4) << 3);
  v8h lo = *(const v8h*)(row);        // K = kb+0..7
  v8h hi = *(const v8h*)(row + 16);   // K = kb+16..23
  return cat8(lo, hi);
}

// B fragment (32x16 f16) from LDS stored [N x ld] (K contiguous per column):
// B[k][n] = L[n*ld + k]; a lane's K-values are two 8-half runs (kb+0..7, kb+16..23) -> 2x b128.
__device__ __forceinline__ v16h load_b_lds(const _Float16* L, int ld,
                                           int k0, int n0, int lane) {
  const _Float16* col = L + (n0 + (lane & 15)) * ld + k0 + ((lane >> 4) << 3);
  v8h lo = *(const v8h*)(col);
  v8h hi = *(const v8h*)(col + 16);
  return cat8(lo, hi);
}

__device__ __forceinline__ v8f wmma16(v16h a, v16h b, v8f c) {
  return __builtin_amdgcn_wmma_f32_16x16x32_f16(false, a, false, b, (short)0, c,
                                                false, false);
}

// ---- prep: convert all weights (fp32) to f16 into d_ws, contiguous segments ----
__global__ __launch_bounds__(256)
void cvt_weights_kernel(const float* __restrict__ tw, const float* __restrict__ pw,
                        const float* __restrict__ gw, const float* __restrict__ ww,
                        _Float16* __restrict__ o) {
  int idx4 = (blockIdx.x * 256 + threadIdx.x) * 4;     // 128 blocks -> 131072 halfs
  int seg  = idx4 >> 15;
  int off  = idx4 & 32767;
  const float* s = (seg == 0) ? tw : (seg == 1) ? pw : (seg == 2) ? gw : ww;
  float4 v = *(const float4*)(s + off);
  v4h h = { (_Float16)v.x, (_Float16)v.y, (_Float16)v.z, (_Float16)v.w };
  *(v4h*)(o + idx4) = h;
}

__global__ __launch_bounds__(256)
void nonlocal_win_kernel(const float* __restrict__ x,
                         const float* __restrict__ theta_b,
                         const float* __restrict__ phi_b,
                         const float* __restrict__ g_b,
                         const float* __restrict__ w_b,
                         const _Float16* __restrict__ Wf16,
                         float* __restrict__ out) {
  extern __shared__ char smem_raw[];
  _Float16* X16T = (_Float16*)smem_raw;          // [64][LDX]  X^T  (s, c)
  _Float16* THT  = X16T + 64 * LDX;              // [64][LDI]  theta^T (s, i)
  _Float16* PHT  = THT + 64 * LDI;               // [64][LDI]  phi^T   (t, i)
  _Float16* GG   = PHT + 64 * LDI;               // [128][LDT] g       (i, t)
  float*    F    = (float*)(GG + 128 * LDT);     // [64][LDF]  logits  (s, t)
  _Float16* ATn  = (_Float16*)(F + 64 * LDF);    // [64][LDT]  attn    (s, t)
  _Float16* Yn   = ATn + 64 * LDT;               // [64][LDI]  y       (s, i)

  const int tid  = threadIdx.x;
  const int lane = tid & 31;
  const int wv   = tid >> 5;
  const int nwin = blockIdx.x;
  const int half8 = ((lane >> 4) << 3);

  // ---- gather window input -> X^T f16 (8-float runs contiguous in x) ----
#pragma unroll
  for (int r = 0; r < 8; ++r) {
    int grp = tid + (r << 8);            // 0..2047 groups of 8
    int j0  = grp << 3;                  // window-flat index, wi == 0
    int cw  = j0 >> 6;
    int s0  = j0 & 63;
    const float4* src = (const float4*)(x + win_to_x(nwin, j0));
    float4 v0 = src[0];
    float4 v1 = src[1];
    _Float16 h[8];
    h[0] = (_Float16)v0.x; h[1] = (_Float16)v0.y;
    h[2] = (_Float16)v0.z; h[3] = (_Float16)v0.w;
    h[4] = (_Float16)v1.x; h[5] = (_Float16)v1.y;
    h[6] = (_Float16)v1.z; h[7] = (_Float16)v1.w;
#pragma unroll
    for (int q = 0; q < 8; ++q) X16T[(s0 + q) * LDX + cw] = h[q];
  }
  __syncthreads();

  // ---- fused theta/phi/g projections: share B tiles of X^T ----
  {
    v8f aT[4], aP[4], aG[4];
#pragma unroll
    for (int nt = 0; nt < 4; ++nt) { aT[nt] = zero8(); aP[nt] = zero8(); aG[nt] = zero8(); }
#pragma unroll
    for (int kt = 0; kt < 8; ++kt) {
      v16h b0 = load_b_lds(X16T, LDX, kt * 32, 0,  lane);
      v16h b1 = load_b_lds(X16T, LDX, kt * 32, 16, lane);
      v16h b2 = load_b_lds(X16T, LDX, kt * 32, 32, lane);
      v16h b3 = load_b_lds(X16T, LDX, kt * 32, 48, lane);
      v16h at_ = load_a16(Wf16 + WOFF_TH, C_INd, wv * 16, kt * 32, lane);
      aT[0] = wmma16(at_, b0, aT[0]); aT[1] = wmma16(at_, b1, aT[1]);
      aT[2] = wmma16(at_, b2, aT[2]); aT[3] = wmma16(at_, b3, aT[3]);
      v16h ap_ = load_a16(Wf16 + WOFF_PH, C_INd, wv * 16, kt * 32, lane);
      aP[0] = wmma16(ap_, b0, aP[0]); aP[1] = wmma16(ap_, b1, aP[1]);
      aP[2] = wmma16(ap_, b2, aP[2]); aP[3] = wmma16(ap_, b3, aP[3]);
      v16h ag_ = load_a16(Wf16 + WOFF_G, C_INd, wv * 16, kt * 32, lane);
      aG[0] = wmma16(ag_, b0, aG[0]); aG[1] = wmma16(ag_, b1, aG[1]);
      aG[2] = wmma16(ag_, b2, aG[2]); aG[3] = wmma16(ag_, b3, aG[3]);
    }
    // biases: 8 consecutive m per lane -> two float4 loads each
    int mb = wv * 16 + half8;
    float bt[8], bp[8], bg[8];
    *(float4*)(bt)     = *(const float4*)(theta_b + mb);
    *(float4*)(bt + 4) = *(const float4*)(theta_b + mb + 4);
    *(float4*)(bp)     = *(const float4*)(phi_b + mb);
    *(float4*)(bp + 4) = *(const float4*)(phi_b + mb + 4);
    *(float4*)(bg)     = *(const float4*)(g_b + mb);
    *(float4*)(bg + 4) = *(const float4*)(g_b + mb + 4);
#pragma unroll
    for (int nt = 0; nt < 4; ++nt) {
#pragma unroll
      for (int r = 0; r < 8; ++r) {
        int m = mb + r;
        int n = nt * 16 + (lane & 15);
        THT[n * LDI + m] = (_Float16)(aT[nt][r] + bt[r]);   // theta^T (s,i)
        PHT[n * LDI + m] = (_Float16)(aP[nt][r] + bp[r]);   // phi^T   (t,i)
        GG[m * LDT + n]  = (_Float16)(aG[nt][r] + bg[r]);   // g       (i,t)
      }
    }
  }
  __syncthreads();

  // ---- f = theta^T @ phi : (64x128)@(128x64); 16 tiles, 2 per wave ----
#pragma unroll
  for (int q = 0; q < 2; ++q) {
    int t  = wv * 2 + q;
    int st = t >> 2, tt = t & 3;
    v8f acc = zero8();
#pragma unroll
    for (int kt = 0; kt < 4; ++kt) {
      v16h a = load_a16(THT, LDI, st * 16, kt * 32, lane);    // A[s][i]
      v16h b = load_b_lds(PHT, LDI, kt * 32, tt * 16, lane);  // B[i][t]
      acc = wmma16(a, b, acc);
    }
#pragma unroll
    for (int r = 0; r < 8; ++r) {
      int s = st * 16 + r + half8;
      int c = tt * 16 + (lane & 15);
      F[s * LDF + c] = acc[r];
    }
  }
  __syncthreads();

  // ---- row softmax (fp32) -> attn f16, natural [s][t] layout ----
  if (tid < 64) {
    const float* row = F + tid * LDF;
    float mx = row[0];
    for (int t = 1; t < 64; ++t) mx = fmaxf(mx, row[t]);
    float sum = 0.0f;
    for (int t = 0; t < 64; ++t) sum += __expf(row[t] - mx);
    float inv = 1.0f / sum;
    for (int t = 0; t < 64; ++t)
      ATn[tid * LDT + t] = (_Float16)(__expf(row[t] - mx) * inv);
  }
  __syncthreads();

  // ---- y^T = g @ attn^T : (128x64)@(64x64); wave wv owns i-tile wv ----
  {
    v8f acc[4];
#pragma unroll
    for (int nt = 0; nt < 4; ++nt) acc[nt] = zero8();
#pragma unroll
    for (int kt = 0; kt < 2; ++kt) {
      v16h a = load_a16(GG, LDT, wv * 16, kt * 32, lane);      // A[i][t]
#pragma unroll
      for (int nt = 0; nt < 4; ++nt) {
        v16h b = load_b_lds(ATn, LDT, kt * 32, nt * 16, lane); // B[t][s]
        acc[nt] = wmma16(a, b, acc[nt]);
      }
    }
#pragma unroll
    for (int nt = 0; nt < 4; ++nt) {
#pragma unroll
      for (int r = 0; r < 8; ++r) {
        int m = wv * 16 + r + half8;                // i
        int n = nt * 16 + (lane & 15);              // s
        Yn[n * LDI + m] = (_Float16)acc[nt][r];     // store y as [s][i]
      }
    }
  }
  __syncthreads();

  // ---- z = w_w @ y^T + bias + residual; share B tiles across the 2 m-tiles ----
  {
    v8f acc[2][4];
#pragma unroll
    for (int mi = 0; mi < 2; ++mi)
#pragma unroll
      for (int nt = 0; nt < 4; ++nt) acc[mi][nt] = zero8();
#pragma unroll
    for (int kt = 0; kt < 4; ++kt) {
      v16h b0 = load_b_lds(Yn, LDI, kt * 32, 0,  lane);
      v16h b1 = load_b_lds(Yn, LDI, kt * 32, 16, lane);
      v16h b2 = load_b_lds(Yn, LDI, kt * 32, 32, lane);
      v16h b3 = load_b_lds(Yn, LDI, kt * 32, 48, lane);
#pragma unroll
      for (int mi = 0; mi < 2; ++mi) {
        v16h a = load_a16(Wf16 + WOFF_W, C_INNERd, (wv * 2 + mi) * 16, kt * 32, lane);
        acc[mi][0] = wmma16(a, b0, acc[mi][0]);
        acc[mi][1] = wmma16(a, b1, acc[mi][1]);
        acc[mi][2] = wmma16(a, b2, acc[mi][2]);
        acc[mi][3] = wmma16(a, b3, acc[mi][3]);
      }
    }
    // Scatter: o[2:0]==r maps to the ti bit-field -> x-address steps by H*W=16384.
#pragma unroll
    for (int mi = 0; mi < 2; ++mi) {
      int o0 = (wv * 2 + mi) * 16 + half8;
      float wb[8];
      *(float4*)(wb)     = *(const float4*)(w_b + o0);
      *(float4*)(wb + 4) = *(const float4*)(w_b + o0 + 4);
#pragma unroll
      for (int nt = 0; nt < 4; ++nt) {
        int s = nt * 16 + (lane & 15);
        int addr = win_to_x(nwin, (o0 << 6) + s);
#pragma unroll
        for (int r = 0; r < 8; ++r) {
          out[addr] = acc[mi][nt][r] + wb[r] + x[addr];   // residual re-read
          addr += 16384;                                  // next o -> ti+1
        }
      }
    }
  }
}

extern "C" void kernel_launch(void* const* d_in, const int* in_sizes, int n_in,
                              void* d_out, int out_size, void* d_ws, size_t ws_size,
                              hipStream_t stream) {
  (void)in_sizes; (void)n_in; (void)out_size; (void)ws_size;
  const float* x       = (const float*)d_in[0];
  const float* theta_w = (const float*)d_in[1];
  const float* theta_b = (const float*)d_in[2];
  const float* phi_w   = (const float*)d_in[3];
  const float* phi_b   = (const float*)d_in[4];
  const float* g_w     = (const float*)d_in[5];
  const float* g_b     = (const float*)d_in[6];
  const float* w_w     = (const float*)d_in[7];
  const float* w_b     = (const float*)d_in[8];
  float* out = (float*)d_out;
  _Float16* wf16 = (_Float16*)d_ws;

  cvt_weights_kernel<<<dim3(W_TOTAL / 4 / 256), dim3(256), 0, stream>>>(
      theta_w, phi_w, g_w, w_w, wf16);
  nonlocal_win_kernel<<<dim3(N_WIN), dim3(256), SMEM_BYTES, stream>>>(
      x, theta_b, phi_b, g_b, w_b, wf16, out);
}
